// GPT_40845138985315
// MI455X (gfx1250) — compile-verified, matrix operand order checked by reading, not run on hardware
//
#include <hip/hip_runtime.h>
#include <math.h>
#include <stdint.h>

// ---------------------------------------------------------------------------
// Types for WMMA (gfx1250, wave32)
// ---------------------------------------------------------------------------
typedef __attribute__((ext_vector_type(16))) __bf16          v16bf;
typedef __attribute__((ext_vector_type(4)))  __bf16          bf16x4;
typedef __attribute__((ext_vector_type(8)))  float           v8f;
typedef __attribute__((ext_vector_type(8)))  unsigned short  ushort8;
typedef __attribute__((ext_vector_type(16))) unsigned short  ushort16;

#define BM 128
#define BN 128
#define BK 64
#define PK 72   // padded LDS row stride (ushorts): 144B (9*16B) -> conflict-free

#define FLAG_ACC  1  // C += result (residual)
#define FLAG_RELU 2  // relu(result)

__device__ __forceinline__ bf16x4 cvt4(float4 v) {
  bf16x4 r;
  r.x = (__bf16)v.x; r.y = (__bf16)v.y; r.z = (__bf16)v.z; r.w = (__bf16)v.w;
  return r;
}

// ---------------------------------------------------------------------------
// Async global->LDS row copy (gfx1250 GLOBAL_LOAD_ASYNC_TO_LDS_B128, ASYNCcnt)
// 256 threads x 16B = 4KB = 1024 floats. Caller must s_wait_asynccnt + barrier.
// ---------------------------------------------------------------------------
__device__ __forceinline__ void async_row_to_lds(float* lds_buf,
                                                 const float* g_row, int tid) {
  unsigned           ldsa = (unsigned)(uintptr_t)lds_buf + (unsigned)tid * 16u;
  unsigned long long ga   = (unsigned long long)(uintptr_t)g_row +
                            (unsigned long long)tid * 16ull;
  asm volatile("global_load_async_to_lds_b128 %0, %1, off"
               :: "v"(ldsa), "v"(ga) : "memory");
}

__device__ __forceinline__ void wait_async_and_barrier() {
  asm volatile("s_wait_asynccnt 0" ::: "memory");
  __syncthreads();
}

// Load one 16x32 bf16 fragment (A-layout) from an LDS row base.
// lane 0-15 : M=lane,   K = kb+khalf+0..7 and +16..23
// lane 16-31: M=lane-16,K = kb+khalf+8..15 and +24..31  (khalf = (lane>>4)*8)
__device__ __forceinline__ v16bf load_frag(const unsigned short* p, int off) {
  ushort8 lo = *reinterpret_cast<const ushort8*>(p + off);
  ushort8 hi = *reinterpret_cast<const ushort8*>(p + off + 16);
  ushort16 w = __builtin_shufflevector(lo, hi, 0,1,2,3,4,5,6,7,8,9,10,11,12,13,14,15);
  return __builtin_bit_cast(v16bf, w);
}

// Stage one 128x64 A tile and one 128x64 B tile (B given as [N,K]) into LDS
// as bf16.  256 threads: thread covers 32 contiguous K of one row.
__device__ __forceinline__ void stage_tiles(
    unsigned short (*As)[PK], unsigned short (*Bs)[PK],
    const float* __restrict__ A, int lda,
    const float* __restrict__ B, int ldb,
    int bm, int bn, int k0, int N, int tid)
{
  const int r = tid >> 1;            // 0..127
  const int c = (tid & 1) * 32;      // 0 or 32
  // ---- A tile (always in-bounds: M,K multiples of tile) ----
  {
    const float* arow = A + (long long)(bm + r) * lda + k0 + c;
    __builtin_prefetch(arow + BK, 0, 3);      // next K tile -> global_prefetch_b8
#pragma unroll
    for (int i = 0; i < 32; i += 4) {
      float4 v = *reinterpret_cast<const float4*>(arow + i);
      *reinterpret_cast<bf16x4*>(&As[r][c + i]) = cvt4(v);
    }
  }
  // ---- B tile: W[N,K] row-major, contiguous along K ----
  {
    const bool ok = (bn + r) < N;
    const float* brow = B + (long long)(bn + r) * ldb + k0 + c;
    if (ok) __builtin_prefetch(brow + BK, 0, 3);
#pragma unroll
    for (int i = 0; i < 32; i += 4) {
      float4 v = ok ? *reinterpret_cast<const float4*>(brow + i)
                    : make_float4(0.f, 0.f, 0.f, 0.f);
      *reinterpret_cast<bf16x4*>(&Bs[r][c + i]) = cvt4(v);
    }
  }
}

// ---------------------------------------------------------------------------
// Batched GEMM: C = A[M,K] x B^T (+C) (relu), B = [N,K] row-major.
// block = 256 threads (8 waves); block tile 128x128; wave tile 32x64.
// Double-buffered LDS, BK=64 (two 16x16x32 WMMA K-steps per stage).
// ---------------------------------------------------------------------------
__global__ void __launch_bounds__(256)
gemm_wmma_kernel(const float* __restrict__ A, int lda, long long sA,
                 const float* __restrict__ B, int ldb, long long sB,
                 float* __restrict__ C, int ldc, long long sC,
                 int M, int N, int K, int flags)
{
  __shared__ __align__(16) unsigned short As[2][BM][PK];
  __shared__ __align__(16) unsigned short Bs[2][BN][PK];

  const int tid  = threadIdx.x;
  const int lane = tid & 31;
  const int wave = tid >> 5;
  const int bm   = blockIdx.x * BM;
  const int bn   = blockIdx.y * BN;

  A += (long long)blockIdx.z * sA;
  B += (long long)blockIdx.z * sB;
  C += (long long)blockIdx.z * sC;

  const int wm = (wave & 3) * 32;   // wave row offset inside block tile
  const int wn = (wave >> 2) * 64;  // wave col offset inside block tile

  v8f acc[2][4] = {};

  const int nK = K / BK;
  stage_tiles(As[0], Bs[0], A, lda, B, ldb, bm, bn, 0, N, tid);

  const int khalf = (lane >> 4) * 8;
  const int fr    = lane & 15;

  int cur = 0;
  for (int kt = 0; kt < nK; ++kt) {
    __syncthreads();
    if (kt + 1 < nK)
      stage_tiles(As[cur ^ 1], Bs[cur ^ 1], A, lda, B, ldb, bm, bn,
                  (kt + 1) * BK, N, tid);

#pragma unroll
    for (int s = 0; s < 2; ++s) {            // two 32-wide WMMA K-steps
      const int kb = s * 32 + khalf;
      v16bf af[2], bfv[4];
#pragma unroll
      for (int mi = 0; mi < 2; ++mi)
        af[mi] = load_frag(&As[cur][wm + mi * 16 + fr][0], kb);
#pragma unroll
      for (int ni = 0; ni < 4; ++ni)
        bfv[ni] = load_frag(&Bs[cur][wn + ni * 16 + fr][0], kb);

#pragma unroll
      for (int mi = 0; mi < 2; ++mi)
#pragma unroll
        for (int ni = 0; ni < 4; ++ni)
          acc[mi][ni] = __builtin_amdgcn_wmma_f32_16x16x32_bf16(
              false, af[mi], false, bfv[ni], (short)0, acc[mi][ni], false, false);
    }
    cur ^= 1;
  }

  // Store (C layout: VGPR r -> M=r (lanes 0-15), M=r+8 (lanes 16-31); N=lane&15)
#pragma unroll
  for (int mi = 0; mi < 2; ++mi) {
#pragma unroll
    for (int ni = 0; ni < 4; ++ni) {
      const int col = bn + wn + ni * 16 + (lane & 15);
      if (col < N) {
#pragma unroll
        for (int r = 0; r < 8; ++r) {
          const int row = bm + wm + mi * 16 + ((lane >> 4) * 8) + r;
          const long long idx = (long long)row * ldc + col;
          float val = acc[mi][ni][r];
          if (flags & FLAG_RELU) val = fmaxf(val, 0.f);
          if (flags & FLAG_ACC)  val += C[idx];
          C[idx] = val;
        }
      }
    }
  }
}

// ---------------------------------------------------------------------------
// Row kernels: async-stage the 4KB row into LDS once, then work out of LDS.
// ---------------------------------------------------------------------------
__global__ void embed_rms_kernel(const int* __restrict__ idx,
                                 const float* __restrict__ wte,
                                 float* __restrict__ x, int D)
{
  __shared__ __align__(16) float rowbuf[1024];
  __shared__ float sm[256];
  const int t = blockIdx.x;
  const float* row = wte + (long long)idx[t] * D;

  async_row_to_lds(rowbuf, row, threadIdx.x);
  wait_async_and_barrier();

  float ss = 0.f;
  for (int d = threadIdx.x; d < D; d += 256) { float v = rowbuf[d]; ss += v * v; }
  sm[threadIdx.x] = ss; __syncthreads();
  for (int s = 128; s > 0; s >>= 1) {
    if (threadIdx.x < s) sm[threadIdx.x] += sm[threadIdx.x + s];
    __syncthreads();
  }
  const float rs = rsqrtf(sm[0] / (float)D + 1e-5f);
  for (int d = threadIdx.x; d < D; d += 256)
    x[(long long)t * D + d] = rowbuf[d] * rs;
}

__global__ void rmsnorm_kernel(const float* __restrict__ in,
                               float* __restrict__ out, int D)
{
  __shared__ __align__(16) float rowbuf[1024];
  __shared__ float sm[256];
  const int t = blockIdx.x;

  async_row_to_lds(rowbuf, in + (long long)t * D, threadIdx.x);
  wait_async_and_barrier();

  float ss = 0.f;
  for (int d = threadIdx.x; d < D; d += 256) { float v = rowbuf[d]; ss += v * v; }
  sm[threadIdx.x] = ss; __syncthreads();
  for (int s = 128; s > 0; s >>= 1) {
    if (threadIdx.x < s) sm[threadIdx.x] += sm[threadIdx.x + s];
    __syncthreads();
  }
  const float rs = rsqrtf(sm[0] / (float)D + 1e-5f);
  for (int d = threadIdx.x; d < D; d += 256)
    out[(long long)t * D + d] = rowbuf[d] * rs;
}

// Build qcat/kcat = [H,T,128] (nope half + RoPE'd half) and vhT = [H,64,T].
__global__ void rope_pack_kernel(const float* __restrict__ q,
                                 const float* __restrict__ qr,
                                 const float* __restrict__ k,
                                 const float* __restrict__ kr,
                                 const float* __restrict__ v,
                                 float* __restrict__ qcat,
                                 float* __restrict__ kcat,
                                 float* __restrict__ vhT, int T, int H)
{
  const int t = blockIdx.x, h = blockIdx.y, j = threadIdx.x;   // j in [0,64)
  const int D = H * 64;
  const long long base = (long long)h * T + t;
  float* qc = qcat + base * 128;
  float* kc = kcat + base * 128;

  qc[j] = q[(long long)t * D + h * 64 + j];
  kc[j] = k[(long long)t * D + h * 64 + j];
  // V stored transposed: vhT[h][j][t]  -> every GEMM uses the [N,K] fast path
  vhT[((long long)h * 64 + j) * T + t] = v[(long long)t * D + h * 64 + j];

  const int i = j & 31;
  const float freq = __powf(10000.f, -((float)(2 * i)) / 64.f);
  const float ang  = (float)t * freq;
  const float c = __cosf(ang), s = __sinf(ang);

  const float qrv = qr[(long long)t * D + h * 64 + j];
  const float qro = (j < 32) ? -qr[(long long)t * D + h * 64 + j + 32]
                             :  qr[(long long)t * D + h * 64 + j - 32];
  qc[64 + j] = qrv * c + qro * s;

  const float krv = kr[(long long)t * 64 + j];
  const float kro = (j < 32) ? -kr[(long long)t * 64 + j + 32]
                             :  kr[(long long)t * 64 + j - 32];
  kc[64 + j] = krv * c + kro * s;
}

// Masked, scaled softmax over the last dim of scores[H,T,T] (T = 1024).
// Row staged into LDS asynchronously; 1 global read + 1 global write total.
__global__ void softmax_kernel(float* __restrict__ scores, int T, float scale)
{
  __shared__ __align__(16) float rowbuf[1024];
  __shared__ float sm[256];
  const int q = blockIdx.x, h = blockIdx.y, tid = threadIdx.x;
  float* row = scores + ((long long)h * T + q) * T;

  async_row_to_lds(rowbuf, row, tid);
  wait_async_and_barrier();

  float m = -3.4e38f;
  for (int k = tid; k < T; k += 256)
    if (k <= q) m = fmaxf(m, rowbuf[k] * scale);
  sm[tid] = m; __syncthreads();
  for (int s = 128; s > 0; s >>= 1) {
    if (tid < s) sm[tid] = fmaxf(sm[tid], sm[tid + s]);
    __syncthreads();
  }
  m = sm[0]; __syncthreads();

  float sum = 0.f;
  for (int k = tid; k < T; k += 256) {
    float p = (k <= q) ? __expf(rowbuf[k] * scale - m) : 0.f;
    rowbuf[k] = p;
    sum += p;
  }
  sm[tid] = sum; __syncthreads();
  for (int s = 128; s > 0; s >>= 1) {
    if (tid < s) sm[tid] += sm[tid + s];
    __syncthreads();
  }
  const float inv = 1.f / sm[0];
  for (int k = tid; k < T; k += 256) row[k] = rowbuf[k] * inv;
}

__global__ void merge_heads_kernel(const float* __restrict__ outh,
                                   float* __restrict__ outm, int T)
{
  const int t = blockIdx.x;
  for (int d = threadIdx.x; d < 1024; d += 256) {
    const int h = d >> 6, j = d & 63;
    outm[(long long)t * 1024 + d] = outh[((long long)h * T + t) * 64 + j];
  }
}

// ---------------------------------------------------------------------------
// Host orchestration
// ---------------------------------------------------------------------------
static void gemm(hipStream_t st,
                 const float* A, int lda, long long sA,
                 const float* B, int ldb, long long sB,
                 float* C, int ldc, long long sC,
                 int M, int N, int K, int batch, int flags)
{
  dim3 grid(M / BM, (N + BN - 1) / BN, batch);
  gemm_wmma_kernel<<<grid, dim3(256), 0, st>>>(A, lda, sA, B, ldb, sB,
                                               C, ldc, sC, M, N, K, flags);
}

extern "C" void kernel_launch(void* const* d_in, const int* in_sizes, int n_in,
                              void* d_out, int out_size, void* d_ws, size_t ws_size,
                              hipStream_t stream)
{
  (void)in_sizes; (void)n_in; (void)out_size; (void)ws_size;
  const int* idx        = (const int*)d_in[0];
  const float* wte      = (const float*)d_in[1];
  const float* wqdown   = (const float*)d_in[2];
  const float* wqup     = (const float*)d_in[3];
  const float* wqr      = (const float*)d_in[4];
  const float* wkvdown  = (const float*)d_in[5];
  const float* wkup     = (const float*)d_in[6];
  const float* wvup     = (const float*)d_in[7];
  const float* wkr      = (const float*)d_in[8];
  const float* wo       = (const float*)d_in[9];
  const float* fc1      = (const float*)d_in[10];
  const float* fc2      = (const float*)d_in[11];
  const float* lm_head  = (const float*)d_in[12];

  const int L = 8, D = 1024, H = 16, LAT = 512, RD = 64, V = 32000, T = 1024;
  const float scale = 1.f / sqrtf(128.f);   // 1/sqrt(hd + RD)
  typedef long long ll;

  // Workspace partition (float elements, 256B-aligned chunks).
  float* ws = (float*)d_ws;
  size_t off = 0;
  auto take = [&](size_t n) { float* p = ws + off; off += (n + 63) & ~(size_t)63; return p; };
  float* x    = take((size_t)T * D);
  float* nx   = take((size_t)T * D);
  float* ql   = take((size_t)T * LAT);
  float* qb   = take((size_t)T * D);
  float* qrb  = take((size_t)T * H * RD);
  float* ckv  = take((size_t)T * LAT);
  float* kb   = take((size_t)T * D);
  float* vb   = take((size_t)T * D);
  float* krb  = take((size_t)T * RD);
  float* qcat = take((size_t)H * T * 128);
  float* kcat = take((size_t)H * T * 128);
  float* vhT  = take((size_t)H * 64 * T);
  float* sc   = take((size_t)H * T * T);
  float* outh = take((size_t)H * T * 64);
  float* outm = take((size_t)T * D);
  float* hb   = take((size_t)T * 4 * D);

  // x = rms_norm(wte[idx])
  embed_rms_kernel<<<T, 256, 0, stream>>>(idx, wte, x, D);

  for (int l = 0; l < L; ++l) {
    const float* Wqd = wqdown + (ll)l * LAT * D;
    const float* Wqu = wqup   + (ll)l * D * LAT;
    const float* Wqr = wqr    + (ll)l * (H * RD) * LAT;
    const float* Wkd = wkvdown+ (ll)l * LAT * D;
    const float* Wku = wkup   + (ll)l * D * LAT;
    const float* Wvu = wvup   + (ll)l * D * LAT;
    const float* Wkr = wkr    + (ll)l * RD * D;
    const float* Wo  = wo     + (ll)l * D * D;
    const float* F1  = fc1    + (ll)l * 4 * D * D;
    const float* F2  = fc2    + (ll)l * D * 4 * D;

    rmsnorm_kernel<<<T, 256, 0, stream>>>(x, nx, D);

    gemm(stream, nx, D, 0, Wqd, D, 0, ql, LAT, 0, T, LAT, D, 1, 0);
    gemm(stream, ql, LAT, 0, Wqu, LAT, 0, qb, D, 0, T, D, LAT, 1, 0);
    gemm(stream, ql, LAT, 0, Wqr, LAT, 0, qrb, H * RD, 0, T, H * RD, LAT, 1, 0);
    gemm(stream, nx, D, 0, Wkd, D, 0, ckv, LAT, 0, T, LAT, D, 1, 0);
    gemm(stream, ckv, LAT, 0, Wku, LAT, 0, kb, D, 0, T, D, LAT, 1, 0);
    gemm(stream, ckv, LAT, 0, Wvu, LAT, 0, vb, D, 0, T, D, LAT, 1, 0);
    gemm(stream, nx, D, 0, Wkr, D, 0, krb, RD, 0, T, RD, D, 1, 0);

    rope_pack_kernel<<<dim3(T, H), 64, 0, stream>>>(qb, qrb, kb, krb, vb,
                                                    qcat, kcat, vhT, T, H);

    // scores[h] = qcat[h] @ kcat[h]^T   (K = 128)
    gemm(stream, qcat, 128, (ll)T * 128, kcat, 128, (ll)T * 128,
         sc, T, (ll)T * T, T, T, 128, H, 0);
    softmax_kernel<<<dim3(T, H), 256, 0, stream>>>(sc, T, scale);
    // outh[h] = attn[h] @ vhT[h]^T      (vhT = [64, T] per head, N = 64)
    gemm(stream, sc, T, (ll)T * T, vhT, T, (ll)64 * T,
         outh, 64, (ll)T * 64, T, 64, T, H, 0);

    merge_heads_kernel<<<T, 256, 0, stream>>>(outh, outm, T);
    // x += out @ wo^T
    gemm(stream, outm, D, 0, Wo, D, 0, x, D, 0, T, D, D, 1, FLAG_ACC);

    // MLP
    rmsnorm_kernel<<<T, 256, 0, stream>>>(x, nx, D);
    gemm(stream, nx, D, 0, F1, D, 0, hb, 4 * D, 0, T, 4 * D, D, 1, FLAG_RELU);
    gemm(stream, hb, 4 * D, 0, F2, 4 * D, 0, x, D, 0, T, D, 4 * D, 1, FLAG_ACC);
  }

  // logits = x @ lm_head^T  -> d_out [T, V]
  gemm(stream, x, D, 0, lm_head, D, 0, (float*)d_out, V, 0, T, V, D, 1, 0);
}
